// SimpleGATModel_87376814670436
// MI455X (gfx1250) — matrix operand order repaired and checked by reading, not compiled, not run on hardware
//
#include <hip/hip_runtime.h>
#include <hip/hip_bf16.h>

// ---- problem constants (fixed by the reference) ----
#define NN   100000      // nodes
#define EE   1600000     // edges
#define FIN  128         // input features
#define EDF  9           // edge features
#define NH   4           // heads
#define NC   16          // channels per head
#define HC   64          // NH*NC
#define NEG_SLOPE 0.2f
#define EPS_F 1e-16f

typedef __attribute__((ext_vector_type(16))) _Float16 v16h;
typedef __attribute__((ext_vector_type(8)))  _Float16 v8h;
typedef __attribute__((ext_vector_type(8)))  float    v8f;

// ---------------------------------------------------------------------------
// ordered-float atomic max via signed-max / unsigned-min trick (init = -inf)
// ---------------------------------------------------------------------------
__device__ __forceinline__ void atomicMaxFloatOrdered(float* addr, float val) {
    if (val >= 0.0f) {
        atomicMax((int*)addr, __float_as_int(val));
    } else {
        atomicMin((unsigned int*)addr, __float_as_uint(val));
    }
}

// ---------------------------------------------------------------------------
// Kernel 0: tiny precompute  Medge[d][h] = sum_c W_edge[d][h*16+c]*att_edge[h][c]
// ---------------------------------------------------------------------------
__global__ void gat_medge_kernel(const float* __restrict__ We,
                                 const float* __restrict__ atte,
                                 float* __restrict__ Medge) {
    int i = blockIdx.x * blockDim.x + threadIdx.x;
    if (i >= EDF * NH) return;
    int d = i >> 2, h = i & 3;
    float s = 0.f;
    #pragma unroll
    for (int c = 0; c < NC; ++c) s += We[d * HC + h * NC + c] * atte[h * NC + c];
    Medge[d * NH + h] = s;
}

// ---------------------------------------------------------------------------
// Kernel 1: init m = -inf, s = 0, agg = 0   (must run every launch: graph replay)
// ---------------------------------------------------------------------------
__global__ void gat_init_kernel(float* __restrict__ m,
                                float* __restrict__ ssum,
                                float* __restrict__ agg) {
    long long i = (long long)blockIdx.x * blockDim.x + threadIdx.x;
    const long long nm = (long long)NN * NH;
    if (i < nm) {
        ((int*)m)[i] = 0xFF800000;                  // -inf
    } else if (i < 2 * nm) {
        ssum[i - nm] = 0.f;
    } else if (i < 2 * nm + (long long)NN * HC) {
        agg[i - 2 * nm] = 0.f;
    }
}

// ---------------------------------------------------------------------------
// Kernel 2: node projection via WMMA (f16 in, f32 acc)
//   xp = x @ W          [N,64]
//   a_src/a_dst via 8 appended B-columns (u-vectors), computed on the fly.
// block = 256 threads (8 waves), block tile = 128 rows.
// ---------------------------------------------------------------------------
__global__ void __launch_bounds__(256)
gat_proj_wmma_kernel(const float* __restrict__ x,
                     const float* __restrict__ W,
                     const float* __restrict__ attsrc,
                     const float* __restrict__ attdst,
                     float* __restrict__ xp,
                     float* __restrict__ a_src,
                     float* __restrict__ a_dst) {
    // LDS: W transposed (cols 0..63) + u-columns (64..71) + zero pad (72..79)
    __shared__ _Float16 sW[80][FIN + 8];    // [col][k], +8 halves pad -> bank spread
    __shared__ _Float16 sX[128][FIN + 8];   // [row][k]

    const int tid     = threadIdx.x;
    const int rowbase = blockIdx.x * 128;

    // --- cooperative fill: W transposed, f32 -> f16 (coalesced global reads) ---
    for (int i = tid; i < FIN * HC; i += 256) {
        int k = i >> 6, c = i & 63;
        sW[c][k] = (_Float16)W[i];
    }
    // --- u-columns: fold att_src/att_dst into 8 extra B columns ---
    if (tid < FIN) {
        int k = tid;
        #pragma unroll
        for (int h = 0; h < NH; ++h) {
            float s1 = 0.f, s2 = 0.f;
            #pragma unroll
            for (int c = 0; c < NC; ++c) {
                float w = W[k * HC + h * NC + c];
                s1 += w * attsrc[h * NC + c];
                s2 += w * attdst[h * NC + c];
            }
            sW[64 + h][k] = (_Float16)s1;
            sW[68 + h][k] = (_Float16)s2;
        }
    }
    for (int i = tid; i < 8 * FIN; i += 256)
        sW[72 + (i >> 7)][i & 127] = (_Float16)0.f;

    // --- x tile: 128 rows, float4 loads, convert to f16 ---
    const float4* x4 = (const float4*)x;    // row stride = 32 float4
    for (int i = tid; i < 128 * 32; i += 256) {
        int row = i >> 5, kq = i & 31;
        int g = rowbase + row;
        float4 v = make_float4(0.f, 0.f, 0.f, 0.f);
        if (g < NN) v = x4[(size_t)g * 32 + kq];
        int k = kq << 2;
        sX[row][k + 0] = (_Float16)v.x;
        sX[row][k + 1] = (_Float16)v.y;
        sX[row][k + 2] = (_Float16)v.z;
        sX[row][k + 3] = (_Float16)v.w;
    }
    __syncthreads();

    const int wave  = tid >> 5;
    const int lane  = tid & 31;
    const int group = lane >> 4;   // lane half
    const int l16   = lane & 15;

    v8f acc[5];
    #pragma unroll
    for (int t = 0; t < 5; ++t) acc[t] = (v8f)(0.f);

    #pragma unroll
    for (int kc = 0; kc < 4; ++kc) {
        // A fragment: 16x32 f16. lane half g: K = kc*32 + {g*8..g*8+7, 16+g*8..23+g*8}
        int rowl = wave * 16 + l16;
        int ka   = kc * 32 + group * 8;
        v8h alo = *(const v8h*)&sX[rowl][ka];
        v8h ahi = *(const v8h*)&sX[rowl][ka + 16];
        v16h afrag;
        #pragma unroll
        for (int i = 0; i < 8; ++i) { afrag[i] = alo[i]; afrag[i + 8] = ahi[i]; }

        #pragma unroll
        for (int nt = 0; nt < 5; ++nt) {
            // B fragment: 32x16 f16. lane half g: K = kc*32 + {g*8..g*8+7, 16+g*8..23+g*8}, col = l16
            int colg = nt * 16 + l16;
            int kb   = kc * 32 + group * 8;
            v8h blo = *(const v8h*)&sW[colg][kb];
            v8h bhi = *(const v8h*)&sW[colg][kb + 16];
            v16h bfrag;
            #pragma unroll
            for (int i = 0; i < 8; ++i) { bfrag[i] = blo[i]; bfrag[i + 8] = bhi[i]; }

            acc[nt] = __builtin_amdgcn_wmma_f32_16x16x32_f16(
                false, afrag, false, bfrag, (short)0, acc[nt], false, false);
        }
    }

    // --- store D: xp tiles (nt 0..3) + a_src/a_dst (nt 4, cols 0..7) ---
    #pragma unroll
    for (int nt = 0; nt < 4; ++nt) {
        int colg = nt * 16 + l16;
        #pragma unroll
        for (int r = 0; r < 8; ++r) {
            int rowg = rowbase + wave * 16 + r + 8 * group;
            if (rowg < NN) xp[(size_t)rowg * HC + colg] = acc[nt][r];
        }
    }
    {
        int j = l16;
        #pragma unroll
        for (int r = 0; r < 8; ++r) {
            int rowg = rowbase + wave * 16 + r + 8 * group;
            if (rowg < NN && j < 8) {
                if (j < 4) a_src[(size_t)rowg * NH + j]       = acc[4][r];
                else       a_dst[(size_t)rowg * NH + (j - 4)] = acc[4][r];
            }
        }
    }
}

// ---------------------------------------------------------------------------
// Kernel 3: per-edge alpha = lrelu(a_src[src] + a_dst[dst] + ea@Medge); seg-max
// ---------------------------------------------------------------------------
__global__ void gat_alpha_kernel(const int* __restrict__ ei,
                                 const float* __restrict__ ea,
                                 const float* __restrict__ Medge,
                                 const float* __restrict__ a_src,
                                 const float* __restrict__ a_dst,
                                 float* __restrict__ alpha,
                                 float* __restrict__ m) {
    long long e = (long long)blockIdx.x * blockDim.x + threadIdx.x;
    if (e >= EE) return;
    int s_ = ei[e];
    int d_ = ei[(long long)EE + e];
    float ah[NH] = {0.f, 0.f, 0.f, 0.f};
    #pragma unroll
    for (int d = 0; d < EDF; ++d) {
        float v = ea[e * EDF + d];
        #pragma unroll
        for (int h = 0; h < NH; ++h) ah[h] += v * Medge[d * NH + h];
    }
    #pragma unroll
    for (int h = 0; h < NH; ++h) {
        float al = a_src[(size_t)s_ * NH + h] + a_dst[(size_t)d_ * NH + h] + ah[h];
        al = (al > 0.f) ? al : NEG_SLOPE * al;          // leaky relu
        alpha[e * NH + h] = al;
        atomicMaxFloatOrdered(&m[(size_t)d_ * NH + h], al);
    }
}

// ---------------------------------------------------------------------------
// Kernel 4: ex = exp(alpha - m[dst]); seg-sum   (alpha buffer overwritten -> ex)
// ---------------------------------------------------------------------------
__global__ void gat_expsum_kernel(const int* __restrict__ ei,
                                  const float* __restrict__ m,
                                  float* __restrict__ alpha,
                                  float* __restrict__ ssum) {
    long long e = (long long)blockIdx.x * blockDim.x + threadIdx.x;
    if (e >= EE) return;
    int d_ = ei[(long long)EE + e];
    #pragma unroll
    for (int h = 0; h < NH; ++h) {
        float ex = __expf(alpha[e * NH + h] - m[(size_t)d_ * NH + h]);
        alpha[e * NH + h] = ex;
        atomicAdd(&ssum[(size_t)d_ * NH + h], ex);
    }
}

// ---------------------------------------------------------------------------
// Kernel 5: att = ex/(s+eps); scatter msg = xp[src]*att into agg[dst]
// 16 lanes per edge -> coalesced 64B gather per head, atomic f32 adds (L2-resident)
// ---------------------------------------------------------------------------
__global__ void gat_scatter_kernel(const int* __restrict__ ei,
                                   const float* __restrict__ ex,
                                   const float* __restrict__ ssum,
                                   const float* __restrict__ xp,
                                   float* __restrict__ agg,
                                   float* __restrict__ out_att) {
    long long gid = (long long)blockIdx.x * blockDim.x + threadIdx.x;
    long long e = gid >> 4;
    int c = (int)(gid & 15);
    if (e >= EE) return;
    int s_ = ei[e];
    int d_ = ei[(long long)EE + e];
    __builtin_prefetch(&xp[(size_t)s_ * HC], 0, 0);     // global_prefetch_b8
    #pragma unroll
    for (int h = 0; h < NH; ++h) {
        float att = ex[e * NH + h] / (ssum[(size_t)d_ * NH + h] + EPS_F);
        if (c == h) out_att[e * NH + h] = att;          // one lane per head stores
        float msg = xp[(size_t)s_ * HC + h * NC + c] * att;
        atomicAdd(&agg[(size_t)d_ * HC + h * NC + c], msg);
    }
}

// ---------------------------------------------------------------------------
// Kernel 6: acc[n][c] = mean_h agg[n][h][c] + bias[c]
// ---------------------------------------------------------------------------
__global__ void gat_finalize_kernel(const float* __restrict__ agg,
                                    const float* __restrict__ bias,
                                    float* __restrict__ out_acc) {
    long long gid = (long long)blockIdx.x * blockDim.x + threadIdx.x;
    if (gid >= (long long)NN * NC) return;
    long long n = gid >> 4;
    int c = (int)(gid & 15);
    float s = 0.f;
    #pragma unroll
    for (int h = 0; h < NH; ++h) s += agg[n * HC + h * NC + c];
    out_acc[gid] = 0.25f * s + bias[c];
}

// ---------------------------------------------------------------------------
// Kernel 7: copy edge_index into output tuple region
// ---------------------------------------------------------------------------
__global__ void gat_eicopy_kernel(const int* __restrict__ ei, void* out, int mode64) {
    long long i = (long long)blockIdx.x * blockDim.x + threadIdx.x;
    if (i >= 2LL * EE) return;
    if (mode64) ((long long*)out)[i] = (long long)ei[i];
    else        ((float*)out)[i]     = (float)ei[i];
}

// ---------------------------------------------------------------------------
extern "C" void kernel_launch(void* const* d_in, const int* in_sizes, int n_in,
                              void* d_out, int out_size, void* d_ws, size_t ws_size,
                              hipStream_t stream) {
    const float* x     = (const float*)d_in[0];
    const int*   ei    = (const int*)d_in[1];
    const float* ea    = (const float*)d_in[2];
    const float* W     = (const float*)d_in[3];
    const float* We    = (const float*)d_in[4];
    const float* atts  = (const float*)d_in[5];
    const float* attd  = (const float*)d_in[6];
    const float* atte  = (const float*)d_in[7];
    const float* bias  = (const float*)d_in[8];

    // workspace layout (floats)
    float* ws    = (float*)d_ws;
    float* xp    = ws;                                  // N*64
    float* a_src = xp    + (size_t)NN * HC;             // N*4
    float* a_dst = a_src + (size_t)NN * NH;             // N*4
    float* m     = a_dst + (size_t)NN * NH;             // N*4
    float* ssum  = m     + (size_t)NN * NH;             // N*4
    float* agg   = ssum  + (size_t)NN * NH;             // N*64
    float* Medge = agg   + (size_t)NN * HC;             // 64 (36 used)
    float* alpha = Medge + 64;                          // E*4 (reused as ex)

    // output tuple layout: [acc N*16][edge_index 2E][att E*4]
    const long long accN = (long long)NN * NC;
    const long long eiN  = 2LL * EE;
    const long long attN = (long long)EE * NH;
    float* out_acc = (float*)d_out;
    void*  out_ei  = (void*)(out_acc + accN);
    float* out_att;
    int mode64;
    if ((long long)out_size >= accN + 2 * eiN + attN) { // edge_index in 8-byte slots
        mode64 = 1;
        out_att = out_acc + accN + 2 * eiN;
    } else {                                            // edge_index in 4-byte slots
        mode64 = 0;
        out_att = out_acc + accN + eiN;
    }

    // 0) fold W_edge @ att_edge
    gat_medge_kernel<<<1, 64, 0, stream>>>(We, atte, Medge);
    // 1) reset m/s/agg (graph replay safe)
    {
        long long tot = 2LL * NN * NH + (long long)NN * HC;
        gat_init_kernel<<<(unsigned)((tot + 255) / 256), 256, 0, stream>>>(m, ssum, agg);
    }
    // 2) WMMA node projection + attention scalars
    gat_proj_wmma_kernel<<<(NN + 127) / 128, 256, 0, stream>>>(x, W, atts, attd,
                                                               xp, a_src, a_dst);
    // 3) edge alpha + segment max
    gat_alpha_kernel<<<(EE + 255) / 256, 256, 0, stream>>>(ei, ea, Medge,
                                                           a_src, a_dst, alpha, m);
    // 4) exp + segment sum
    gat_expsum_kernel<<<(EE + 255) / 256, 256, 0, stream>>>(ei, m, alpha, ssum);
    // 5) normalize + scatter messages (16 lanes/edge)
    {
        long long tot = (long long)EE * 16;
        gat_scatter_kernel<<<(unsigned)((tot + 255) / 256), 256, 0, stream>>>(
            ei, alpha, ssum, xp, agg, out_att);
    }
    // 6) head mean + bias
    gat_finalize_kernel<<<(unsigned)((accN + 255) / 256), 256, 0, stream>>>(agg, bias,
                                                                            out_acc);
    // 7) edge_index into tuple output
    gat_eicopy_kernel<<<(unsigned)((eiN + 255) / 256), 256, 0, stream>>>(ei, out_ei,
                                                                         mode64);
}
